// CrossAttention3D_6734508720462
// MI455X (gfx1250) — matrix operand order repaired and checked by reading, not compiled, not run on hardware
//
#include <hip/hip_runtime.h>
#include <hip/hip_bf16.h>

typedef __bf16 bf16;
typedef __attribute__((ext_vector_type(16))) __bf16 v16bf;
typedef __attribute__((ext_vector_type(8)))  __bf16 v8bf;
typedef __attribute__((ext_vector_type(8)))  float  v8f;

#define NB 4
#define CC 512
#define SS 4096
#define DD 512

static __device__ __forceinline__ bf16 f2bf(float f) {
  unsigned u = __builtin_bit_cast(unsigned, f);
  unsigned r = u + 0x7FFFu + ((u >> 16) & 1u);
  unsigned short h = (unsigned short)(r >> 16);
  return __builtin_bit_cast(bf16, h);
}

static __device__ __forceinline__ v16bf ldfrag(const bf16* p0, const bf16* p1) {
  v8bf lo = *(const v8bf*)p0;
  v8bf hi = *(const v8bf*)p1;
  return __builtin_shufflevector(lo, hi, 0,1,2,3,4,5,6,7,8,9,10,11,12,13,14,15);
}

// Async global -> LDS copy of one 16B chunk per lane (CDNA5, ASYNCcnt-tracked).
static __device__ __forceinline__ void async_b128(const bf16* gptr, const bf16* lptr) {
  unsigned lds = (unsigned)(unsigned long long)lptr;      // LDS addr = low 32 bits
  unsigned long long ga = (unsigned long long)gptr;
  asm volatile("global_load_async_to_lds_b128 %0, %1, off"
               :: "v"(lds), "v"(ga) : "memory");
}
static __device__ __forceinline__ void wait_async_hex10() {
  asm volatile("s_wait_asynccnt 0x10" ::: "memory");      // allow 16 newest in flight
}
static __device__ __forceinline__ void wait_async_0() {
  asm volatile("s_wait_asynccnt 0x0" ::: "memory");
}

// ---------------------------------------------------------------------------
// Generic projection GEMM:  Out[m][n] = sum_k A[m][k] * W[n][k] + bias[n]
//   AMODE 0: A is f32, stored [K][M] per batch (x case: transpose-on-load)
//   AMODE 1: A is bf16, stored [M][K] per batch (attention-output case)
//   OMODE 0: bf16 Out[m*N+n], value scaled by `scale` (Q: fold softmax scale)
//   OMODE 1: bf16 Out[n*M+m] (transposed store -> V^T, t-contiguous)
//   OMODE 2: f32  Out[n*M+m] (final y[b][c][s] into d_out)
// ---------------------------------------------------------------------------
template<int AMODE, int OMODE>
__global__ __launch_bounds__(256) void proj_gemm(
    const void* __restrict__ Ap, const float* __restrict__ Wt,
    const float* __restrict__ bias, void* __restrict__ Outp,
    float scale, int M, int N, int K, long aBatch, long oBatch)
{
  __shared__ bf16 As[128][40];   // [m][k], k-contiguous (+8 pad)
  __shared__ bf16 Bs[128][40];   // [n][k], k-contiguous (+8 pad)

  const int tid = threadIdx.x;
  const int z  = blockIdx.z;
  const int m0 = blockIdx.x * 128;
  const int n0 = blockIdx.y * 128;
  const int wv = tid >> 5, lane = tid & 31;
  const int wm = wv >> 1, wn = wv & 1;        // 4 waves along M, 2 along N
  const int lm = lane & 15, kg = lane >> 4;

  v8f acc[2][4];
  #pragma unroll
  for (int i = 0; i < 2; i++)
    #pragma unroll
    for (int j = 0; j < 4; j++) acc[i][j] = (v8f)0.0f;

  for (int k0 = 0; k0 < K; k0 += 32) {
    __syncthreads();
    if (AMODE == 0) {
      const float* A = (const float*)Ap + (long)z * aBatch;
      #pragma unroll
      for (int i = 0; i < 16; i++) {
        int idx = tid + i * 256;
        int kl = idx >> 7, ml = idx & 127;      // coalesced along m
        As[ml][kl] = f2bf(A[(long)(k0 + kl) * M + m0 + ml]);
      }
    } else {
      const bf16* A = (const bf16*)Ap + (long)z * aBatch;
      int row = tid >> 1, half = tid & 1;
      const bf16* src = A + (long)(m0 + row) * K + k0 + half * 16;
      v8bf a0 = *(const v8bf*)(src);
      v8bf a1 = *(const v8bf*)(src + 8);
      *(v8bf*)&As[row][half * 16]     = a0;
      *(v8bf*)&As[row][half * 16 + 8] = a1;
    }
    #pragma unroll
    for (int i = 0; i < 16; i++) {
      int idx = tid + i * 256;
      int kl = idx & 31, nl = idx >> 5;         // coalesced along k
      Bs[nl][kl] = f2bf(Wt[(long)(n0 + nl) * K + k0 + kl]);
    }
    __syncthreads();

    v16bf af[2], bfm[4];
    #pragma unroll
    for (int mi = 0; mi < 2; mi++) {
      int r = wm * 32 + mi * 16 + lm;
      af[mi] = ldfrag(&As[r][kg * 8], &As[r][16 + kg * 8]);
    }
    #pragma unroll
    for (int ni = 0; ni < 4; ni++) {
      int c = wn * 64 + ni * 16 + lm;
      bfm[ni] = ldfrag(&Bs[c][kg * 8], &Bs[c][16 + kg * 8]);
    }
    #pragma unroll
    for (int mi = 0; mi < 2; mi++)
      #pragma unroll
      for (int ni = 0; ni < 4; ni++)
        acc[mi][ni] = __builtin_amdgcn_wmma_f32_16x16x32_bf16(
            false, af[mi], false, bfm[ni], (short)0, acc[mi][ni], false, false);
  }

  float bn[4];
  #pragma unroll
  for (int ni = 0; ni < 4; ni++) bn[ni] = bias[n0 + wn * 64 + ni * 16 + lm];

  #pragma unroll
  for (int mi = 0; mi < 2; mi++) {
    #pragma unroll
    for (int ni = 0; ni < 4; ni++) {
      int n = n0 + wn * 64 + ni * 16 + lm;
      int mbase = m0 + wm * 32 + mi * 16 + kg * 8;   // rows v+8*kg, v=0..7
      if (OMODE == 0) {
        bf16* Out = (bf16*)Outp + (long)z * oBatch;
        #pragma unroll
        for (int v = 0; v < 8; v++)
          Out[(long)(mbase + v) * N + n] = f2bf((acc[mi][ni][v] + bn[ni]) * scale);
      } else if (OMODE == 1) {
        bf16* Out = (bf16*)Outp + (long)z * oBatch;
        v8bf pk;
        #pragma unroll
        for (int v = 0; v < 8; v++) pk[v] = f2bf(acc[mi][ni][v] + bn[ni]);
        *(v8bf*)&Out[(long)n * M + mbase] = pk;      // 16B packed store
      } else {
        float* Out = (float*)Outp + (long)z * oBatch;
        #pragma unroll
        for (int v = 0; v < 8; v++)
          Out[(long)n * M + mbase + v] = acc[mi][ni][v] + bn[ni];
      }
    }
  }
}

// ---------------------------------------------------------------------------
// Flash attention: 64 Q rows per workgroup, key tiles of 64, online softmax.
// Q (resident), K and V^T tiles live in LDS (~217KB of the WGP's 320KB).
// K/V staging uses CDNA5 async global->LDS copies, software-pipelined so
// K(t+1) overlaps softmax+PV(t) and V(t+1) overlaps sim(t+1).
// Q pre-scaled by D^-0.5. O overwrites this workgroup's own Q rows.
// ---------------------------------------------------------------------------
__global__ __launch_bounds__(256, 1) void flash_attn(
    const bf16* __restrict__ Qh, const bf16* __restrict__ Kh,
    const bf16* __restrict__ Vt, bf16* __restrict__ Oo)
{
  __shared__ bf16 Ql[64][520];                // Q tile  [s][d], d-contiguous (+8)
  __shared__ bf16 Kl[64][520];                // K tile  [t][d], d-contiguous (+8)
  __shared__ bf16 Vl[512][72];                // V^T tile [d][t], t-contiguous (+8)
  __shared__ bf16 Pl[64][72];                 // P tile  [s][t], t-contiguous (+8)
  __shared__ float mrow[64], lrow[64], arow[64];
  __shared__ float pmax[2][64], psum[2][64];

  const int tid = threadIdx.x;
  const int z  = blockIdx.y;
  const int s0 = blockIdx.x * 64;
  const int wv = tid >> 5, lane = tid & 31;
  const int mg = wv >> 1, ng = wv & 1;        // 4 row-groups x 2 col-groups
  const int lm = lane & 15, kg = lane >> 4;

  const bf16* qb = Qh + (long)z * SS * DD;
  const bf16* kb = Kh + (long)z * SS * DD;
  const bf16* vb = Vt + (long)z * (long)DD * SS;
  bf16*       ob = Oo + (long)z * SS * DD;

  v8f oacc[16];                                // 16 rows x 256 d-cols, f32
  #pragma unroll
  for (int i = 0; i < 16; i++) oacc[i] = (v8f)0.0f;

  // ---- prologue: async-stage Q tile, K(0), V^T(0) ----
  #pragma unroll
  for (int j = 0; j < 16; j++) {               // Q: 4096 x 16B chunks
    int id = tid + j * 256;
    int t = id >> 6, dc = id & 63;
    async_b128(qb + (long)(s0 + t) * DD + dc * 8, &Ql[t][dc * 8]);
  }
  #pragma unroll
  for (int j = 0; j < 16; j++) {               // K(0)
    int id = tid + j * 256;
    int t = id >> 6, dc = id & 63;
    async_b128(kb + (long)t * DD + dc * 8, &Kl[t][dc * 8]);
  }
  #pragma unroll
  for (int j = 0; j < 16; j++) {               // V(0)
    int id = tid + j * 256;
    int d = id >> 3, tc = id & 7;
    async_b128(vb + (long)d * SS + tc * 8, &Vl[d][tc * 8]);
  }
  wait_async_0();
  if (tid < 64) { mrow[tid] = -1e30f; lrow[tid] = 0.0f; }
  __syncthreads();

  for (int t0 = 0; t0 < SS; t0 += 64) {
    const bool more = (t0 + 64) < SS;

    // ---- sim tile = Q . K^T (wave: 16 rows x 32 key-cols), all from LDS ----
    v8f sacc[2] = { (v8f)0.0f, (v8f)0.0f };
    const int srl = mg * 16 + lm;
    #pragma unroll 4
    for (int kc = 0; kc < DD; kc += 32) {
      v16bf a = ldfrag(&Ql[srl][kc + kg * 8], &Ql[srl][kc + 16 + kg * 8]);
      #pragma unroll
      for (int ni = 0; ni < 2; ni++) {
        const bf16* bp = &Kl[ng * 32 + ni * 16 + lm][kc + kg * 8];
        v16bf b = ldfrag(bp, bp + 16);
        sacc[ni] = __builtin_amdgcn_wmma_f32_16x16x32_bf16(
            false, a, false, b, (short)0, sacc[ni], false, false);
      }
    }

    // ---- per-row max (wave-local 32 cols, xor-shuffle over 16 lanes) ----
    #pragma unroll
    for (int v = 0; v < 8; v++) {
      float x = fmaxf(sacc[0][v], sacc[1][v]);
      x = fmaxf(x, __shfl_xor(x, 1, 32));
      x = fmaxf(x, __shfl_xor(x, 2, 32));
      x = fmaxf(x, __shfl_xor(x, 4, 32));
      x = fmaxf(x, __shfl_xor(x, 8, 32));
      if (lm == 0) pmax[ng][mg * 16 + kg * 8 + v] = x;
    }
    __syncthreads();                            // B1: Kl free for next tile

    if (more) {                                 // async K(t+1): overlaps softmax+PV
      #pragma unroll
      for (int j = 0; j < 16; j++) {
        int id = tid + j * 256;
        int t = id >> 6, dc = id & 63;
        async_b128(kb + (long)(t0 + 64 + t) * DD + dc * 8, &Kl[t][dc * 8]);
      }
    }
    if (tid < 64) {
      float mn = fmaxf(mrow[tid], fmaxf(pmax[0][tid], pmax[1][tid]));
      arow[tid] = __expf(mrow[tid] - mn);
      mrow[tid] = mn;
    }
    __syncthreads();                            // B2

    // ---- P = exp(S - m) -> LDS (bf16), plus per-row partial sums ----
    #pragma unroll
    for (int v = 0; v < 8; v++) {
      int r = mg * 16 + kg * 8 + v;
      float mv = mrow[r];
      float p0 = __expf(sacc[0][v] - mv);
      float p1 = __expf(sacc[1][v] - mv);
      Pl[r][ng * 32 + lm]      = f2bf(p0);
      Pl[r][ng * 32 + 16 + lm] = f2bf(p1);
      float s = p0 + p1;
      s += __shfl_xor(s, 1, 32);
      s += __shfl_xor(s, 2, 32);
      s += __shfl_xor(s, 4, 32);
      s += __shfl_xor(s, 8, 32);
      if (lm == 0) psum[ng][r] = s;
    }
    if (more) wait_async_hex10(); else wait_async_0();  // V(t0) arrived (in-order)
    __syncthreads();                            // B3: Pl + everyone's V chunks ready

    if (tid < 64) lrow[tid] = lrow[tid] * arow[tid] + psum[0][tid] + psum[1][tid];

    // ---- rescale running O by alpha ----
    float aloc[8];
    #pragma unroll
    for (int v = 0; v < 8; v++) aloc[v] = arow[mg * 16 + kg * 8 + v];
    #pragma unroll
    for (int ni = 0; ni < 16; ni++)
      #pragma unroll
      for (int v = 0; v < 8; v++) oacc[ni][v] *= aloc[v];

    // ---- O += P . V   (both operands from LDS) ----
    #pragma unroll
    for (int ks = 0; ks < 2; ks++) {
      int r = mg * 16 + lm;
      v16bf a = ldfrag(&Pl[r][ks * 32 + kg * 8], &Pl[r][ks * 32 + 16 + kg * 8]);
      #pragma unroll
      for (int ni = 0; ni < 16; ni++) {
        const bf16* bp = &Vl[ng * 256 + ni * 16 + lm][ks * 32 + kg * 8];
        v16bf b = ldfrag(bp, bp + 16);
        oacc[ni] = __builtin_amdgcn_wmma_f32_16x16x32_bf16(
            false, a, false, b, (short)0, oacc[ni], false, false);
      }
    }
    __syncthreads();                            // B4: Vl free for next tile

    if (more) {                                 // async V(t+1): overlaps sim(t+1)
      #pragma unroll
      for (int j = 0; j < 16; j++) {
        int id = tid + j * 256;
        int d = id >> 3, tc = id & 7;
        async_b128(vb + (long)d * SS + t0 + 64 + tc * 8, &Vl[d][tc * 8]);
      }
      wait_async_hex10();                       // K(t+1) drained (older group)
      __syncthreads();                          // B5: everyone's K chunks visible
    }
  }

  // ---- finalize: O / l, store bf16 (overwrites own Q rows) ----
  __syncthreads();
  float linv[8];
  #pragma unroll
  for (int v = 0; v < 8; v++) linv[v] = 1.0f / lrow[mg * 16 + kg * 8 + v];
  #pragma unroll
  for (int ni = 0; ni < 16; ni++) {
    int d = ng * 256 + ni * 16 + lm;
    #pragma unroll
    for (int v = 0; v < 8; v++) {
      int srw = s0 + mg * 16 + kg * 8 + v;
      ob[(long)srw * DD + d] = f2bf(oacc[ni][v] * linv[v]);
    }
  }
}

// ---------------------------------------------------------------------------
extern "C" void kernel_launch(void* const* d_in, const int* in_sizes, int n_in,
                              void* d_out, int out_size, void* d_ws, size_t ws_size,
                              hipStream_t stream)
{
  (void)in_sizes; (void)n_in; (void)out_size; (void)ws_size;
  const float* x  = (const float*)d_in[0];
  const float* Wq = (const float*)d_in[1];
  const float* bq = (const float*)d_in[2];
  const float* Wk = (const float*)d_in[3];
  const float* bk = (const float*)d_in[4];
  const float* Wv = (const float*)d_in[5];
  const float* bv = (const float*)d_in[6];
  const float* Wo = (const float*)d_in[7];
  const float* bo = (const float*)d_in[8];

  bf16* Qh = (bf16*)d_ws;                 // also reused as attention output O
  bf16* Kh = Qh + (long)NB * SS * DD;
  bf16* Vt = Kh + (long)NB * SS * DD;     // V transposed: [b][d][t]

  const float scale = 0.04419417382415922f;  // 512^-0.5, folded into Q
  dim3 blk(256);
  dim3 gproj(SS / 128, DD / 128, NB);

  proj_gemm<0, 0><<<gproj, blk, 0, stream>>>(x, Wq, bq, Qh, scale,
      SS, DD, CC, (long)CC * SS, (long)SS * DD);
  proj_gemm<0, 0><<<gproj, blk, 0, stream>>>(x, Wk, bk, Kh, 1.0f,
      SS, DD, CC, (long)CC * SS, (long)SS * DD);
  proj_gemm<0, 1><<<gproj, blk, 0, stream>>>(x, Wv, bv, Vt, 1.0f,
      SS, DD, CC, (long)CC * SS, (long)DD * SS);

  flash_attn<<<dim3(SS / 64, NB), blk, 0, stream>>>(Qh, Kh, Vt, Qh);

  proj_gemm<1, 2><<<dim3(SS / 128, CC / 128, NB), blk, 0, stream>>>(Qh, Wo, bo,
      d_out, 1.0f, SS, CC, DD, (long)SS * DD, (long)CC * SS);
}
